// MultiHeadAttention_35914516529799
// MI455X (gfx1250) — matrix operand order repaired and checked by reading, not compiled, not run on hardware
//
#include <hip/hip_runtime.h>
#include <hip/hip_bf16.h>
#include <math.h>

// ---------------------------------------------------------------------------
// MHA forward for MI455X (gfx1250), wave32 + v_wmma_f32_16x16x32_f16.
// Shapes: B=8, C=4 (BC=32), N=1024, D=96, H=3, DK=DV=32.
// ---------------------------------------------------------------------------

typedef _Float16 half_t;
typedef __attribute__((ext_vector_type(16))) _Float16 v16h;
typedef __attribute__((ext_vector_type(8)))  _Float16 v8h;
typedef __attribute__((ext_vector_type(8)))  float    v8f;

#define NTOK   1024
#define DMODEL 96
#define NHEAD  3
#define DHEAD  32
#define NBC    32              // B*C
#define QSCALE 0.17677669529663687f   // 1/sqrt(32)

#define WMMA(a, b, c) \
  __builtin_amdgcn_wmma_f32_16x16x32_f16(false, (a), false, (b), (short)0, (c), false, false)

__device__ __forceinline__ v16h cat8(v8h lo, v8h hi) {
  return __builtin_shufflevector(lo, hi, 0,1,2,3,4,5,6,7,8,9,10,11,12,13,14,15);
}

// A operand (16x32 f16, row-major source). rowp points at this lane's row (M = lane&15).
// K layout per lane: {sel..sel+7, 16+sel..23+sel}, sel = 8*(lane>=16).
__device__ __forceinline__ v16h loadA_f16(const half_t* __restrict__ rowp, int hi) {
  const half_t* p = rowp + hi * 8;
  v8h a = *(const v8h*)(p);
  v8h b = *(const v8h*)(p + 16);
  return cat8(a, b);
}

// Same A layout for an LDS-resident tile.
__device__ __forceinline__ v16h loadA_lds(const half_t* rowp, int hi) {
  const half_t* p = rowp + hi * 8;
  v8h a = *(const v8h*)(p);
  v8h b = *(const v8h*)(p + 16);
  return cat8(a, b);
}

// Same A layout, converting from f32 on the fly.
__device__ __forceinline__ v16h loadA_f32(const float* __restrict__ rowp, int hi) {
  const float* p = rowp + hi * 8;
  v16h r;
#pragma unroll
  for (int i = 0; i < 8; ++i) {
    r[i]     = (half_t)p[i];
    r[8 + i] = (half_t)p[16 + i];
  }
  return r;
}

// B operand (32x16 f16). Lane holds column N = lane&15; K values
// {sel..sel+7, 16+sel..23+sel}, sel = 8*(lane>=16). base points at (row K=0) of this lane's column data.
__device__ __forceinline__ v16h loadB_f16(const half_t* __restrict__ base, int hi) {
  const half_t* p = base + hi * 8;
  v8h a = *(const v8h*)(p);
  v8h b = *(const v8h*)(p + 16);
  return cat8(a, b);
}

// C/D tile (16x16 f32) store to row-major f16: M = r + 8*hi, N = lo.
__device__ __forceinline__ void storeC_f16(half_t* __restrict__ base, int ld,
                                           int lo, int hi, v8f c, float scale) {
#pragma unroll
  for (int r = 0; r < 8; ++r)
    base[(size_t)(r + 8 * hi) * ld + lo] = (half_t)(c[r] * scale);
}

// ---------------------------------------------------------------------------
// Kernel 0: convert all weights f32 -> f16 (each 96x96, row-major).
// ---------------------------------------------------------------------------
__global__ void k_cvt_w(const float* __restrict__ wq, const float* __restrict__ wk,
                        const float* __restrict__ wv, const float* __restrict__ wfc,
                        half_t* __restrict__ oq, half_t* __restrict__ ok,
                        half_t* __restrict__ ov, half_t* __restrict__ ofc) {
  int i = blockIdx.x * blockDim.x + threadIdx.x;
  if (i < DMODEL * DMODEL) {
    oq[i]  = (half_t)wq[i];
    ok[i]  = (half_t)wk[i];
    ov[i]  = (half_t)wv[i];
    ofc[i] = (half_t)wfc[i];
  }
}

// ---------------------------------------------------------------------------
// Kernel 1: projections. One wave per (bc, 16-token tile).
//   Qp[bc][n][96] f16 (pre-scaled by 1/sqrt(dk)), Kp[bc][n][96] f16,
//   Vt[bc][h][32][1024] f16 (transposed so attn@V B-operand is contiguous).
// Straight-line per projection (templated) so every pointer is kernarg-derived
// and loads lower to global_load_* (not flat_load_*).
// ---------------------------------------------------------------------------
template <int MODE>   // 0 = Q (scaled), 1 = K, 2 = V (transposed store)
__device__ __forceinline__ void proj_one(const float* __restrict__ X,
                                         const half_t* __restrict__ W,
                                         half_t* __restrict__ dst,
                                         int bc, int t0, int lo, int hi) {
  const float* xrow = X + ((size_t)bc * NTOK + t0 + lo) * DMODEL;
  v16h A0 = loadA_f32(xrow,      hi);
  v16h A1 = loadA_f32(xrow + 32, hi);
  v16h A2 = loadA_f32(xrow + 64, hi);
#pragma unroll
  for (int eb = 0; eb < 6; ++eb) {
    const half_t* wr = W + (size_t)(eb * 16 + lo) * DMODEL;
    v8f acc = {};
    acc = WMMA(A0, loadB_f16(wr,      hi), acc);
    acc = WMMA(A1, loadB_f16(wr + 32, hi), acc);
    acc = WMMA(A2, loadB_f16(wr + 64, hi), acc);
    if (MODE == 0) {
      storeC_f16(dst + ((size_t)bc * NTOK + t0) * DMODEL + eb * 16,
                 DMODEL, lo, hi, acc, QSCALE);
    } else if (MODE == 1) {
      storeC_f16(dst + ((size_t)bc * NTOK + t0) * DMODEL + eb * 16,
                 DMODEL, lo, hi, acc, 1.0f);
    } else {
      int e = eb * 16 + lo, h = e >> 5, dd = e & 31;
      half_t* vp = dst + ((size_t)(bc * NHEAD + h) * DHEAD + dd) * NTOK + t0;
#pragma unroll
      for (int r = 0; r < 8; ++r) vp[r + 8 * hi] = (half_t)acc[r];
    }
  }
}

__global__ __launch_bounds__(32)
void k_proj(const float* __restrict__ XQ, const float* __restrict__ XK,
            const float* __restrict__ XV, const half_t* __restrict__ Wq,
            const half_t* __restrict__ Wk, const half_t* __restrict__ Wv,
            half_t* __restrict__ Qp, half_t* __restrict__ Kp,
            half_t* __restrict__ Vt) {
  const int lane = threadIdx.x;
  const int lo = lane & 15, hi = lane >> 4;
  const int bc = blockIdx.x >> 6;
  const int t0 = (blockIdx.x & 63) << 4;

  proj_one<0>(XQ, Wq, Qp, bc, t0, lo, hi);
  proj_one<1>(XK, Wk, Kp, bc, t0, lo, hi);
  proj_one<2>(XV, Wv, Vt, bc, t0, lo, hi);
}

// ---------------------------------------------------------------------------
// Kernel 2: column softmax stats (softmax is over the QUERY axis).
// One wave per (bc, h, 16 key-columns). Flash-style running max/sum over q.
// Column reduction: 8 rows in-lane + partner rows via __shfl_xor(.,16).
// ---------------------------------------------------------------------------
__global__ __launch_bounds__(32)
void k_stats(const half_t* __restrict__ Qp, const half_t* __restrict__ Kp,
             float* __restrict__ Mk, float* __restrict__ Zinv) {
  const int lane = threadIdx.x;
  const int lo = lane & 15, hi = lane >> 4;
  const int kt  = blockIdx.x & 63;
  const int bch = blockIdx.x >> 6;          // bc*3 + h
  const int h   = bch % NHEAD, bc = bch / NHEAD;
  const int k0  = kt << 4;

  const half_t* qbase = Qp + (size_t)bc * NTOK * DMODEL + h * DHEAD;
  const half_t* krow  = Kp + ((size_t)bc * NTOK + k0 + lo) * DMODEL + h * DHEAD;
  v16h Bk = loadB_f16(krow, hi);            // fixed key tile (16 columns)

  float m = -1.0e30f, z = 0.0f;
  for (int qt = 0; qt < 64; ++qt) {
    const half_t* arow = qbase + (size_t)(qt * 16 + lo) * DMODEL;
    if (qt < 63)
      __builtin_prefetch(qbase + (size_t)((qt + 1) * 16 + lo) * DMODEL, 0, 0);
    v16h Aq = loadA_f16(arow, hi);
    v8f zero = {};
    v8f s = WMMA(Aq, Bk, zero);             // S[q, k0+lo], 16 q rows split over pair
    float tm = s[0];
#pragma unroll
    for (int r = 1; r < 8; ++r) tm = fmaxf(tm, s[r]);
    tm = fmaxf(tm, __shfl_xor(tm, 16, 32));
    float nm = fmaxf(m, tm);
    float p = 0.0f;
#pragma unroll
    for (int r = 0; r < 8; ++r) p += __expf(s[r] - nm);
    p += __shfl_xor(p, 16, 32);
    z = z * __expf(m - nm) + p;
    m = nm;
  }
  if (hi == 0) {
    size_t idx = (size_t)bch * NTOK + k0 + lo;
    Mk[idx]   = m;
    Zinv[idx] = 1.0f / z;
  }
}

// ---------------------------------------------------------------------------
// Kernel 3: context = P @ V with P[q,k] = exp(S-m_k)*invZ_k (f16).
// One wave per (bc, h, 16-query tile). P staged through LDS to convert the
// D-layout score tile into an A-operand layout.
// ---------------------------------------------------------------------------
__global__ __launch_bounds__(32)
void k_ctx(const half_t* __restrict__ Qp, const half_t* __restrict__ Kp,
           const half_t* __restrict__ Vt, const float* __restrict__ Mk,
           const float* __restrict__ Zinv, half_t* __restrict__ Ctx) {
  __shared__ __align__(16) half_t Pt[16 * 32];

  const int lane = threadIdx.x;
  const int lo = lane & 15, hi = lane >> 4;
  const int qt  = blockIdx.x & 63;
  const int bch = blockIdx.x >> 6;
  const int h   = bch % NHEAD, bc = bch / NHEAD;
  const int q0  = qt << 4;

  const half_t* qrow  = Qp + ((size_t)bc * NTOK + q0 + lo) * DMODEL + h * DHEAD;
  const v16h Aq = loadA_f16(qrow, hi);
  const half_t* kbase = Kp + (size_t)bc * NTOK * DMODEL + h * DHEAD;
  const half_t* vbase = Vt + (size_t)bch * DHEAD * NTOK;
  const float*  mrow  = Mk   + (size_t)bch * NTOK;
  const float*  zrow  = Zinv + (size_t)bch * NTOK;

  v8f acc0 = {}, acc1 = {};
  for (int k = 0; k < NTOK; k += 32) {
    v16h Bk0 = loadB_f16(kbase + (size_t)(k + lo) * DMODEL, hi);
    v16h Bk1 = loadB_f16(kbase + (size_t)(k + 16 + lo) * DMODEL, hi);
    v8f zero = {};
    v8f s0 = WMMA(Aq, Bk0, zero);
    v8f s1 = WMMA(Aq, Bk1, zero);
    float m0 = mrow[k + lo],      iz0 = zrow[k + lo];
    float m1 = mrow[k + 16 + lo], iz1 = zrow[k + 16 + lo];
#pragma unroll
    for (int r = 0; r < 8; ++r) {
      int row = r + 8 * hi;
      Pt[row * 32 + lo]      = (half_t)(__expf(s0[r] - m0) * iz0);
      Pt[row * 32 + 16 + lo] = (half_t)(__expf(s1[r] - m1) * iz1);
    }
    __syncthreads();
    v16h Ap  = loadA_lds(&Pt[lo * 32], hi);                 // P as A operand (LDS)
    v16h Bv0 = loadB_f16(vbase + (size_t)lo * NTOK + k, hi);        // d = 0..15
    v16h Bv1 = loadB_f16(vbase + (size_t)(16 + lo) * NTOK + k, hi); // d = 16..31
    acc0 = WMMA(Ap, Bv0, acc0);
    acc1 = WMMA(Ap, Bv1, acc1);
    __syncthreads();
  }
  half_t* cb = Ctx + ((size_t)bc * NTOK + q0) * DMODEL + h * DHEAD;
  storeC_f16(cb,      DMODEL, lo, hi, acc0, 1.0f);
  storeC_f16(cb + 16, DMODEL, lo, hi, acc1, 1.0f);
}

// ---------------------------------------------------------------------------
// Kernel 4: out = LayerNorm(Ctx @ W_fc^T + input_Q). One wave per 16 tokens.
// ---------------------------------------------------------------------------
__global__ __launch_bounds__(32)
void k_fc_ln(const half_t* __restrict__ Ctx, const half_t* __restrict__ Wfc,
             const float* __restrict__ resid, const float* __restrict__ gamma,
             const float* __restrict__ beta, float* __restrict__ out) {
  __shared__ __align__(16) float Tile[16 * DMODEL];
  __shared__ float mu[16], rs[16];

  const int lane = threadIdx.x;
  const int lo = lane & 15, hi = lane >> 4;
  const int bc = blockIdx.x >> 6;
  const int t0 = (blockIdx.x & 63) << 4;

  const half_t* crow = Ctx + ((size_t)bc * NTOK + t0 + lo) * DMODEL;
  v16h A0 = loadA_f16(crow,      hi);
  v16h A1 = loadA_f16(crow + 32, hi);
  v16h A2 = loadA_f16(crow + 64, hi);
  const float* rb = resid + ((size_t)bc * NTOK + t0) * DMODEL;

#pragma unroll
  for (int db = 0; db < 6; ++db) {
    const half_t* wr = Wfc + (size_t)(db * 16 + lo) * DMODEL;
    v8f acc = {};
    acc = WMMA(A0, loadB_f16(wr,      hi), acc);
    acc = WMMA(A1, loadB_f16(wr + 32, hi), acc);
    acc = WMMA(A2, loadB_f16(wr + 64, hi), acc);
#pragma unroll
    for (int r = 0; r < 8; ++r) {
      int mrow = r + 8 * hi;
      Tile[mrow * DMODEL + db * 16 + lo] =
          acc[r] + rb[(size_t)mrow * DMODEL + db * 16 + lo];
    }
  }
  __syncthreads();
  if (lane < 16) {
    float s = 0.0f, ss = 0.0f;
    for (int d = 0; d < DMODEL; ++d) {
      float v = Tile[lane * DMODEL + d];
      s += v; ss += v * v;
    }
    float mean = s * (1.0f / DMODEL);
    float var  = ss * (1.0f / DMODEL) - mean * mean;
    mu[lane] = mean;
    rs[lane] = rsqrtf(var + 1e-5f);
  }
  __syncthreads();
  float* ob = out + ((size_t)bc * NTOK + t0) * DMODEL;
  for (int idx = lane; idx < 16 * DMODEL; idx += 32) {
    int t = idx / DMODEL, d = idx % DMODEL;
    ob[idx] = (Tile[idx] - mu[t]) * rs[t] * gamma[d] + beta[d];
  }
}

// ---------------------------------------------------------------------------
// Launch
// ---------------------------------------------------------------------------
extern "C" void kernel_launch(void* const* d_in, const int* in_sizes, int n_in,
                              void* d_out, int out_size, void* d_ws, size_t ws_size,
                              hipStream_t stream) {
  (void)in_sizes; (void)n_in; (void)out_size; (void)ws_size;
  const float* XQ    = (const float*)d_in[0];
  const float* XK    = (const float*)d_in[1];
  const float* XV    = (const float*)d_in[2];
  const float* Wq32  = (const float*)d_in[3];
  const float* Wk32  = (const float*)d_in[4];
  const float* Wv32  = (const float*)d_in[5];
  const float* Wfc32 = (const float*)d_in[6];
  const float* gamma = (const float*)d_in[7];
  const float* beta  = (const float*)d_in[8];
  float* out = (float*)d_out;

  // Workspace carve-up (all 256B aligned).
  char* ws = (char*)d_ws;
  size_t off = 0;
  auto alloc = [&](size_t bytes) { void* p = ws + off; off = (off + bytes + 255) & ~(size_t)255; return p; };
  half_t* Wq16  = (half_t*)alloc(DMODEL * DMODEL * 2);
  half_t* Wk16  = (half_t*)alloc(DMODEL * DMODEL * 2);
  half_t* Wv16  = (half_t*)alloc(DMODEL * DMODEL * 2);
  half_t* Wfc16 = (half_t*)alloc(DMODEL * DMODEL * 2);
  half_t* Qp    = (half_t*)alloc((size_t)NBC * NTOK * DMODEL * 2);
  half_t* Kp    = (half_t*)alloc((size_t)NBC * NTOK * DMODEL * 2);
  half_t* Vt    = (half_t*)alloc((size_t)NBC * NHEAD * DHEAD * NTOK * 2);
  float*  Mk    = (float*) alloc((size_t)NBC * NHEAD * NTOK * 4);
  float*  Zinv  = (float*) alloc((size_t)NBC * NHEAD * NTOK * 4);
  half_t* Ctx   = (half_t*)alloc((size_t)NBC * NTOK * DMODEL * 2);

  k_cvt_w<<<(DMODEL * DMODEL + 255) / 256, 256, 0, stream>>>(
      Wq32, Wk32, Wv32, Wfc32, Wq16, Wk16, Wv16, Wfc16);

  k_proj<<<NBC * (NTOK / 16), 32, 0, stream>>>(
      XQ, XK, XV, Wq16, Wk16, Wv16, Qp, Kp, Vt);

  k_stats<<<NBC * NHEAD * (NTOK / 16), 32, 0, stream>>>(Qp, Kp, Mk, Zinv);

  k_ctx<<<NBC * NHEAD * (NTOK / 16), 32, 0, stream>>>(Qp, Kp, Vt, Mk, Zinv, Ctx);

  k_fc_ln<<<NBC * (NTOK / 16), 32, 0, stream>>>(Ctx, Wfc16, XQ, gamma, beta, out);
}
